// uniform_loss_16269336117248
// MI455X (gfx1250) — matrix operand order repaired and checked by reading, not compiled, not run on hardware
//
#include <hip/hip_runtime.h>

// ---------------------------------------------------------------------------
// uniform_loss:  mean_i logsumexp_j( <x_i, x_j> / T ),  x: [16384, 128] f32,
// rows L2-normalized.  <x_i,x_j> <= 1 with equality at j==i, so the row max
// of the logit is exactly 1/T and:
//     lse_i = 1/T + log( sum_j exp( (d_ij - 1)/T ) )     (all exp args <= 0)
// Compute-bound Gram matmul -> bf16 hi/lo split on v_wmma_f32_16x16x32_bf16,
// with double-buffered GLOBAL_LOAD_ASYNC_TO_LDS (no staging VGPRs, no spills).
// ---------------------------------------------------------------------------

#define NROWS 16384
#define NDIM  128
#define LDS_STRIDE 136            // 128 bf16 + 8 pad -> conflict-free b128 reads

static constexpr float INV_T = 1.0f / 0.07f;
// exp((d-1)/T) = 2^(d*K2 - K2),  K2 = log2(e)/T
static constexpr float K2    = 1.4426950408889634f / 0.07f;

typedef __attribute__((ext_vector_type(16))) __bf16 v16bf;
typedef __attribute__((ext_vector_type(8)))  float  v8f;
typedef __attribute__((ext_vector_type(4)))  int    v4i;

// Pointer types matching the async-DMA builtin's signature (from the clang
// diagnostic: param0 is a 16-byte int vector pointer in the global AS).
typedef __attribute__((address_space(1))) v4i* as1_v4i_ptr;
typedef __attribute__((address_space(3))) v4i* as3_v4i_ptr;

#if __has_builtin(__builtin_amdgcn_global_load_async_to_lds_b128)
#define HAVE_ASYNC_LDS 1
#else
#define HAVE_ASYNC_LDS 0
#endif

union Frag {
    v16bf v;
    uint4 q[2];
};

// ---- bit-level fp32 <-> bf16 (round-to-nearest-even) -----------------------
__device__ __forceinline__ __bf16 f32_to_bf16(float f) {
    union { float f; unsigned u; } in; in.f = f;
    unsigned r = in.u + 0x7FFFu + ((in.u >> 16) & 1u);
    union { unsigned short s; __bf16 b; } out; out.s = (unsigned short)(r >> 16);
    return out.b;
}
__device__ __forceinline__ float bf16_to_f32(__bf16 b) {
    union { unsigned short s; __bf16 b; } in; in.b = b;
    union { unsigned u; float f; } out; out.u = ((unsigned)in.s) << 16;
    return out.f;
}

#if HAVE_ASYNC_LDS
__device__ __forceinline__ void wait_asynccnt0() {
#if __has_builtin(__builtin_amdgcn_s_wait_asynccnt)
    __builtin_amdgcn_s_wait_asynccnt(0);
#else
    asm volatile("s_wait_asynccnt 0" ::: "memory");
#endif
}
#endif

// ---------------------------------------------------------------------------
// Kernel 1: split fp32 x into bf16 hi + bf16 residual lo.
// ---------------------------------------------------------------------------
__global__ void split_bf16_kernel(const float* __restrict__ x,
                                  __bf16* __restrict__ xhi,
                                  __bf16* __restrict__ xlo) {
    int i = blockIdx.x * blockDim.x + threadIdx.x;   // grid covers N*D exactly
    float f = x[i];
    __bf16 h = f32_to_bf16(f);
    xhi[i] = h;
    xlo[i] = f32_to_bf16(f - bf16_to_f32(h));
}

// ---------------------------------------------------------------------------
// Kernel 2: fused Gram x WMMA x exp-sum.
//   grid.x = N/128 workgroups, 256 threads = 8 wave32 per WG.
//   Each wave owns a 16-row tile; WG streams 128-column blocks through a
//   double-buffered LDS filled by async global->LDS DMA.
// ---------------------------------------------------------------------------
__global__ void __launch_bounds__(256)
gram_expsum_kernel(const __bf16* __restrict__ xhi,
                   const __bf16* __restrict__ xlo,
                   float* __restrict__ rowsum) {
#if HAVE_ASYNC_LDS
    __shared__ __bf16 lds_hi[2][128 * LDS_STRIDE];   // 2 x 34 KB
    __shared__ __bf16 lds_lo[2][128 * LDS_STRIDE];   // 2 x 34 KB
#else
    __shared__ __bf16 lds_hi[1][128 * LDS_STRIDE];
    __shared__ __bf16 lds_lo[1][128 * LDS_STRIDE];
#endif

    const int tid  = threadIdx.x;
    const int wave = tid >> 5;
    const int lane = tid & 31;
    const int half = lane >> 4;      // 16-lane half of the wave
    const int ml   = lane & 15;      // row (A) / column (B) index within tile

    const int rowBase = blockIdx.x * 128 + wave * 16;

    // ---- A fragments (16 rows x 128 K), resident in VGPRs for whole kernel.
    // 16-bit A layout (ISA 7.12.2): lanes 0-15 hold K {0-7,16-23}+kb,
    // lanes 16-31 hold K {8-15,24-31}+kb  -> two contiguous 16B chunks/lane.
    Frag a_hi[4], a_lo[4];
    {
        const __bf16* arow_hi = xhi + (size_t)(rowBase + ml) * NDIM;
        const __bf16* arow_lo = xlo + (size_t)(rowBase + ml) * NDIM;
#pragma unroll
        for (int ks = 0; ks < 4; ++ks) {
            const int kb = ks * 32;
            const int k0 = kb + 8 * half;        // frag elements 0..7
            const int k1 = kb + 16 + 8 * half;   // frag elements 8..15
            a_hi[ks].q[0] = *(const uint4*)(arow_hi + k0);
            a_hi[ks].q[1] = *(const uint4*)(arow_hi + k1);
            a_lo[ks].q[0] = *(const uint4*)(arow_lo + k0);
            a_lo[ks].q[1] = *(const uint4*)(arow_lo + k1);
        }
    }

    // Per-lane partial sums: acc[v] belongs to row (rowBase + half*8 + v)
    // per the 16x16 f32 C/D layout.
    float acc[8];
#pragma unroll
    for (int v = 0; v < 8; ++v) acc[v] = 0.0f;

    constexpr int NB = NROWS / 128;

#if HAVE_ASYNC_LDS
    // Per-thread chunk coordinates: 8 chunks per array, 16-row stride.
    const int r0  = tid >> 4;            // 0..15
    const int c8a = (tid & 15) * 8;      // bf16 column of this thread's b128
    auto issue_block = [&](int cb, int buf) {
        const size_t gbase = (size_t)cb * 128 * NDIM;
#pragma unroll
        for (int i = 0; i < 8; ++i) {
            const int r = r0 + i * 16;
            const __bf16* gh = xhi + gbase + (size_t)r * NDIM + c8a;
            const __bf16* gl = xlo + gbase + (size_t)r * NDIM + c8a;
            __bf16* lh = &lds_hi[buf][r * LDS_STRIDE + c8a];
            __bf16* ll = &lds_lo[buf][r * LDS_STRIDE + c8a];
            __builtin_amdgcn_global_load_async_to_lds_b128(
                (as1_v4i_ptr)gh, (as3_v4i_ptr)lh, 0, 0);
            __builtin_amdgcn_global_load_async_to_lds_b128(
                (as1_v4i_ptr)gl, (as3_v4i_ptr)ll, 0, 0);
        }
    };

    issue_block(0, 0);
    wait_asynccnt0();
    __syncthreads();
#endif

    for (int cb = 0; cb < NB; ++cb) {
#if HAVE_ASYNC_LDS
        const int cur = cb & 1;
        // DMA the next block into the other buffer; flies under the WMMAs.
        if (cb + 1 < NB) issue_block(cb + 1, cur ^ 1);
#else
        const int cur = 0;
        {   // synchronous cooperative stage
            const size_t gbase = (size_t)cb * 128 * NDIM;
#pragma unroll
            for (int i = 0; i < 8; ++i) {
                const int idx = tid + i * 256;
                const int r   = idx >> 4;
                const int c8  = (idx & 15) * 8;
                uint4 h = *(const uint4*)(xhi + gbase + (size_t)r * NDIM + c8);
                uint4 l = *(const uint4*)(xlo + gbase + (size_t)r * NDIM + c8);
                *(uint4*)(&lds_hi[0][r * LDS_STRIDE + c8]) = h;
                *(uint4*)(&lds_lo[0][r * LDS_STRIDE + c8]) = l;
            }
        }
        __syncthreads();
#endif

        // ---- 8 sub-tiles of 16 columns each -------------------------------
        for (int st = 0; st < 8; ++st) {
            const int ncol = st * 16 + ml;
            // 16-bit B layout (32x16), matching A: lanes 0-15 hold K {0-7,16-23},
            // lanes 16-31 hold K {8-15,24-31} -> two 16B chunks per lane per k-step.
            const __bf16* brow_hi = &lds_hi[cur][ncol * LDS_STRIDE + 8 * half];
            const __bf16* brow_lo = &lds_lo[cur][ncol * LDS_STRIDE + 8 * half];
            Frag b_hi[4], b_lo[4];
#pragma unroll
            for (int ks = 0; ks < 4; ++ks) {
                b_hi[ks].q[0] = *(const uint4*)(brow_hi + ks * 32);
                b_hi[ks].q[1] = *(const uint4*)(brow_hi + ks * 32 + 16);
                b_lo[ks].q[0] = *(const uint4*)(brow_lo + ks * 32);
                b_lo[ks].q[1] = *(const uint4*)(brow_lo + ks * 32 + 16);
            }

            // Two independent accumulator chains (6 WMMAs each) to halve the
            // D->C serialization on the matrix pipe.
            v8f c0 = {0.f, 0.f, 0.f, 0.f, 0.f, 0.f, 0.f, 0.f};
            v8f c1 = {0.f, 0.f, 0.f, 0.f, 0.f, 0.f, 0.f, 0.f};
            c0 = __builtin_amdgcn_wmma_f32_16x16x32_bf16(false, a_hi[0].v, false, b_hi[0].v, (short)0, c0, false, false);
            c1 = __builtin_amdgcn_wmma_f32_16x16x32_bf16(false, a_hi[1].v, false, b_hi[1].v, (short)0, c1, false, false);
            c0 = __builtin_amdgcn_wmma_f32_16x16x32_bf16(false, a_hi[2].v, false, b_hi[2].v, (short)0, c0, false, false);
            c1 = __builtin_amdgcn_wmma_f32_16x16x32_bf16(false, a_hi[3].v, false, b_hi[3].v, (short)0, c1, false, false);
            c0 = __builtin_amdgcn_wmma_f32_16x16x32_bf16(false, a_hi[0].v, false, b_lo[0].v, (short)0, c0, false, false);
            c1 = __builtin_amdgcn_wmma_f32_16x16x32_bf16(false, a_hi[1].v, false, b_lo[1].v, (short)0, c1, false, false);
            c0 = __builtin_amdgcn_wmma_f32_16x16x32_bf16(false, a_hi[2].v, false, b_lo[2].v, (short)0, c0, false, false);
            c1 = __builtin_amdgcn_wmma_f32_16x16x32_bf16(false, a_hi[3].v, false, b_lo[3].v, (short)0, c1, false, false);
            c0 = __builtin_amdgcn_wmma_f32_16x16x32_bf16(false, a_lo[0].v, false, b_hi[0].v, (short)0, c0, false, false);
            c1 = __builtin_amdgcn_wmma_f32_16x16x32_bf16(false, a_lo[1].v, false, b_hi[1].v, (short)0, c1, false, false);
            c0 = __builtin_amdgcn_wmma_f32_16x16x32_bf16(false, a_lo[2].v, false, b_hi[2].v, (short)0, c0, false, false);
            c1 = __builtin_amdgcn_wmma_f32_16x16x32_bf16(false, a_lo[3].v, false, b_hi[3].v, (short)0, c1, false, false);

            // exp((d-1)/T) = 2^(d*K2 - K2): one v_fma_f32 + one v_exp_f32.
            // Argument <= 0 -> no overflow; trans ops co-execute with WMMA.
#pragma unroll
            for (int v = 0; v < 8; ++v) {
                float d = c0[v] + c1[v];
                acc[v] += __builtin_amdgcn_exp2f(__builtin_fmaf(d, K2, -K2));
            }
        }

#if HAVE_ASYNC_LDS
        wait_asynccnt0();    // next buffer's DMA complete (per-wave ASYNCcnt)
        __syncthreads();     // everyone done reading cur, sees next filled
#else
        __syncthreads();
#endif
    }

    // ---- reduce each row's 16 column-partials across its 16-lane half -----
#pragma unroll
    for (int v = 0; v < 8; ++v) {
        float s = acc[v];
        s += __shfl_xor(s, 1, 32);
        s += __shfl_xor(s, 2, 32);
        s += __shfl_xor(s, 4, 32);
        s += __shfl_xor(s, 8, 32);
        if (ml == v) rowsum[rowBase + half * 8 + v] = s;
    }
}

// ---------------------------------------------------------------------------
// Kernel 3: out = 1/T + mean_i log(rowsum_i)
// ---------------------------------------------------------------------------
__global__ void lse_mean_kernel(const float* __restrict__ rowsum,
                                float* __restrict__ out) {
    __shared__ float sm[256];
    const int tid = threadIdx.x;
    float s = 0.0f;
    for (int i = tid; i < NROWS; i += 256)
        s += logf(rowsum[i]);
    sm[tid] = s;
    __syncthreads();
    for (int w = 128; w > 0; w >>= 1) {
        if (tid < w) sm[tid] += sm[tid + w];
        __syncthreads();
    }
    if (tid == 0)
        out[0] = INV_T + sm[0] / (float)NROWS;
}

// ---------------------------------------------------------------------------
extern "C" void kernel_launch(void* const* d_in, const int* in_sizes, int n_in,
                              void* d_out, int out_size, void* d_ws, size_t ws_size,
                              hipStream_t stream) {
    const float* x = (const float*)d_in[0];

    // workspace: xhi [N*D] bf16 | xlo [N*D] bf16 | rowsum [N] f32  (~8.4 MB)
    __bf16* xhi    = (__bf16*)d_ws;
    __bf16* xlo    = xhi + (size_t)NROWS * NDIM;
    float*  rowsum = (float*)(xlo + (size_t)NROWS * NDIM);
    float*  out    = (float*)d_out;

    split_bf16_kernel<<<(NROWS * NDIM) / 256, 256, 0, stream>>>(x, xhi, xlo);
    gram_expsum_kernel<<<NROWS / 128, 256, 0, stream>>>(xhi, xlo, rowsum);
    lse_mean_kernel<<<1, 256, 0, stream>>>(rowsum, out);
}
